// TableCellFetcher_5849745457732
// MI455X (gfx1250) — matrix operand order repaired and checked by reading, not compiled, not run on hardware
//
#include <hip/hip_runtime.h>
#include <hip/hip_bf16.h>

#define B_  8
#define L_  1024
#define D_  512
#define H_  8
#define DH_ 64
#define DF_ 2048

typedef __attribute__((ext_vector_type(16))) _Float16 v16h;
typedef __attribute__((ext_vector_type(8)))  _Float16 v8h;
typedef __attribute__((ext_vector_type(8)))  float    v8f;

// ---------------------------------------------------------------------------
// soc mask + stable compaction order (one thread per batch; L=1024 loop)
// ---------------------------------------------------------------------------
__global__ void mask_order_kernel(const int* __restrict__ seq,
                                  const int* __restrict__ SOCp,
                                  const int* __restrict__ EOSp,
                                  int* __restrict__ order,
                                  int* __restrict__ counts) {
    int b = threadIdx.x;
    if (b >= B_) return;
    int soc = SOCp[0], eos = EOSp[0];
    int cnt = 0, seenEos = 0;
    for (int i = 0; i < L_; ++i) {
        int s = seq[b * L_ + i];
        if (s == eos) seenEos = 1;                 // cumsum(seq==EOS) semantics
        if (s == soc && !seenEos) order[b * L_ + cnt++] = i;
    }
    counts[b] = cnt;
}

// ---------------------------------------------------------------------------
// ext = gather(hid, order), zero padded rows  (one block per row)
// ---------------------------------------------------------------------------
__global__ void gather_rows_kernel(const float* __restrict__ hid,
                                   const int* __restrict__ order,
                                   const int* __restrict__ counts,
                                   float* __restrict__ x) {
    int row = blockIdx.x;           // b*L + i
    int b = row / L_, i = row % L_;
    float* d = x + (long)row * D_;
    if (i < counts[b]) {
        const float* s = hid + ((long)b * L_ + order[b * L_ + i]) * D_;
        for (int c = threadIdx.x; c < D_; c += blockDim.x) d[c] = s[c];
    } else {
        for (int c = threadIdx.x; c < D_; c += blockDim.x) d[c] = 0.0f;
    }
}

// ---------------------------------------------------------------------------
// elementwise f32 -> f16
// ---------------------------------------------------------------------------
__global__ void cvt_f16_kernel(const float* __restrict__ in,
                               _Float16* __restrict__ out, long n) {
    long stride = (long)gridDim.x * blockDim.x;
    for (long i = (long)blockIdx.x * blockDim.x + threadIdx.x; i < n; i += stride)
        out[i] = (_Float16)in[i];
}

// ---------------------------------------------------------------------------
// W [K,N] f32 -> WT [N,K] f16  (B-operand layout for WMMA GEMM)
// ---------------------------------------------------------------------------
__global__ void transpose_w_kernel(const float* __restrict__ W,
                                   _Float16* __restrict__ WT, int K, int N) {
    long n = (long)K * N;
    long stride = (long)gridDim.x * blockDim.x;
    for (long i = (long)blockIdx.x * blockDim.x + threadIdx.x; i < n; i += stride) {
        int k = (int)(i / N), c = (int)(i % N);
        WT[(long)c * K + k] = (_Float16)W[i];
    }
}

// ---------------------------------------------------------------------------
// v f16 [B, L, H*DH] -> vT f16 [B, H, DH, L]  (so o-GEMM B-operand is [N=DH,K=L])
// ---------------------------------------------------------------------------
__global__ void transpose_v_kernel(const _Float16* __restrict__ v,
                                   _Float16* __restrict__ vT) {
    long n = (long)B_ * L_ * D_;
    long stride = (long)gridDim.x * blockDim.x;
    for (long i = (long)blockIdx.x * blockDim.x + threadIdx.x; i < n; i += stride) {
        int b = (int)(i / ((long)L_ * D_));
        long rem = i % ((long)L_ * D_);
        int l = (int)(rem / D_), c = (int)(rem % D_);
        int h = c / DH_, d = c % DH_;
        vT[(((long)b * H_ + h) * DH_ + d) * L_ + l] = v[i];
    }
}

// ---------------------------------------------------------------------------
// row LayerNorm (D=512), f32 in -> f16 out  (one 256-thread block per row)
// ---------------------------------------------------------------------------
__global__ void ln_f16_kernel(const float* __restrict__ x,
                              _Float16* __restrict__ out) {
    int row = blockIdx.x;
    const float* xr = x + (long)row * D_;
    __shared__ float s1[256], s2[256];
    int t = threadIdx.x;
    float a = xr[t], b2 = xr[t + 256];
    s1[t] = a + b2;
    s2[t] = a * a + b2 * b2;
    __syncthreads();
    for (int o = 128; o > 0; o >>= 1) {
        if (t < o) { s1[t] += s1[t + o]; s2[t] += s2[t + o]; }
        __syncthreads();
    }
    float mean = s1[0] * (1.0f / D_);
    float var  = s2[0] * (1.0f / D_) - mean * mean;
    float r = rsqrtf(var + 1e-5f);
    out[(long)row * D_ + t]       = (_Float16)((a  - mean) * r);
    out[(long)row * D_ + t + 256] = (_Float16)((b2 - mean) * r);
}

// ---------------------------------------------------------------------------
// masked softmax over rows of [B,H,L,L] f16, in place.
// counts!=null: key c valid iff c < counts[b] (== -1e9 masking in reference).
// ---------------------------------------------------------------------------
__global__ void softmax_kernel(_Float16* __restrict__ probs,
                               const int* __restrict__ counts) {
    int rid = blockIdx.x;                   // 0 .. B*H*L-1
    int b = rid / (H_ * L_);
    int limit = counts ? counts[b] : L_;
    _Float16* row = probs + (long)rid * L_;
    int t = threadIdx.x;
    __shared__ float red[256];

    float v[4], mx = -1e30f;
    #pragma unroll
    for (int j = 0; j < 4; ++j) {
        int c = t + j * 256;
        v[j] = (float)row[c];
        if (c < limit) mx = fmaxf(mx, v[j]);
    }
    red[t] = mx; __syncthreads();
    for (int o = 128; o > 0; o >>= 1) { if (t < o) red[t] = fmaxf(red[t], red[t + o]); __syncthreads(); }
    mx = red[0];
    __syncthreads();

    float e[4], sum = 0.0f;
    #pragma unroll
    for (int j = 0; j < 4; ++j) {
        int c = t + j * 256;
        e[j] = (c < limit) ? __expf(v[j] - mx) : 0.0f;
        sum += e[j];
    }
    red[t] = sum; __syncthreads();
    for (int o = 128; o > 0; o >>= 1) { if (t < o) red[t] += red[t + o]; __syncthreads(); }
    sum = red[0];

    if (limit > 0) {
        float inv = 1.0f / sum;
        #pragma unroll
        for (int j = 0; j < 4; ++j) row[t + j * 256] = (_Float16)(e[j] * inv);
    } else {
        #pragma unroll
        for (int j = 0; j < 4; ++j) row[t + j * 256] = (_Float16)(1.0f / L_);
    }
}

// ---------------------------------------------------------------------------
// WMMA fragment loads (contiguous k order, same for A and B)
// A 16x32 f16: lane<16 -> K[0..15]; lane>=16 -> K[16..31], contiguous
// ---------------------------------------------------------------------------
__device__ __forceinline__ v16h load_a_frag(const _Float16* p) {
    v8h lo = *(const v8h*)p;          // K base + half*16 .. +7
    v8h hi = *(const v8h*)(p + 8);    // K base + half*16 + 8 .. +7
    return __builtin_shufflevector(lo, hi, 0,1,2,3,4,5,6,7,8,9,10,11,12,13,14,15);
}
// B 32x16 f16 (stored as BT[N,K]): lane<16 -> K[0..15]; lane>=16 -> K[16..31], contiguous
__device__ __forceinline__ v16h load_b_frag(const _Float16* p) {
    v8h lo = *(const v8h*)p;
    v8h hi = *(const v8h*)(p + 8);
    return __builtin_shufflevector(lo, hi, 0,1,2,3,4,5,6,7,8,9,10,11,12,13,14,15);
}

// ---------------------------------------------------------------------------
// Batched WMMA GEMM:  C[z] (+)= scale * A[z] @ BT[z]^T   (BT stored [N,K])
// batch offset = (z/innerCnt)*Outer + (z%innerCnt)*Inner for each operand.
// Block = 256 threads = 8 waves; block tile 128(M) x 64(N); wave tile 16x64.
// Requires M%128==0, N%64==0, K%32==0 (true for all uses here).
// ---------------------------------------------------------------------------
__global__ __launch_bounds__(256)
void wmma_gemm_kernel(const _Float16* __restrict__ A, long aO, long aI,
                      const _Float16* __restrict__ Bt, long bO, long bI,
                      float* __restrict__ Cf, _Float16* __restrict__ Ch,
                      long cO, long cI,
                      int innerCnt, int lda, int ldb, int ldc,
                      int K, float scale, int accumulate, int gelu) {
    int z  = blockIdx.z;
    int zo = z / innerCnt, zi = z % innerCnt;
    const _Float16* Ab = A  + zo * aO + zi * aI;
    const _Float16* Bb = Bt + zo * bO + zi * bI;
    long cOff = zo * cO + zi * cI;

    int wave = threadIdx.x >> 5;
    int lane = threadIdx.x & 31;
    int lr   = lane & 15;
    int half = lane >> 4;

    int m0 = blockIdx.y * 128 + wave * 16;
    int n0 = blockIdx.x * 64;

    v8f acc[4] = {};

    const _Float16* arow = Ab + (long)(m0 + lr) * lda + half * 16;
    const _Float16* brow0 = Bb + (long)(n0 +  0 + lr) * ldb + half * 16;
    const _Float16* brow1 = Bb + (long)(n0 + 16 + lr) * ldb + half * 16;
    const _Float16* brow2 = Bb + (long)(n0 + 32 + lr) * ldb + half * 16;
    const _Float16* brow3 = Bb + (long)(n0 + 48 + lr) * ldb + half * 16;

    for (int k0 = 0; k0 < K; k0 += 32) {
        v16h a = load_a_frag(arow + k0);
        acc[0] = __builtin_amdgcn_wmma_f32_16x16x32_f16(false, a, false, load_b_frag(brow0 + k0), (short)0, acc[0], false, false);
        acc[1] = __builtin_amdgcn_wmma_f32_16x16x32_f16(false, a, false, load_b_frag(brow1 + k0), (short)0, acc[1], false, false);
        acc[2] = __builtin_amdgcn_wmma_f32_16x16x32_f16(false, a, false, load_b_frag(brow2 + k0), (short)0, acc[2], false, false);
        acc[3] = __builtin_amdgcn_wmma_f32_16x16x32_f16(false, a, false, load_b_frag(brow3 + k0), (short)0, acc[3], false, false);
    }

    // C/D layout: VGPR j -> (m = m0 + half*8 + j, n = n0 + t*16 + lr)
    #pragma unroll
    for (int t2 = 0; t2 < 4; ++t2) {
        int n = n0 + t2 * 16 + lr;
        #pragma unroll
        for (int j = 0; j < 8; ++j) {
            int m = m0 + half * 8 + j;
            float val = acc[t2][j] * scale;
            if (gelu) {
                float u = val;
                val = 0.5f * u * (1.0f + tanhf(0.7978845608f * (u + 0.044715f * u * u * u)));
            }
            long idx = cOff + (long)m * ldc + n;
            if (accumulate) val += Cf[idx];
            if (Ch) Ch[idx] = (_Float16)val;
            else    Cf[idx] = val;
        }
    }
}

// ---------------------------------------------------------------------------
// out[0..n) = hid ; out[n..2n) = x
// ---------------------------------------------------------------------------
__global__ void copy_out_kernel(const float* __restrict__ hid,
                                const float* __restrict__ x,
                                float* __restrict__ out, long n) {
    long stride = (long)gridDim.x * blockDim.x;
    for (long i = (long)blockIdx.x * blockDim.x + threadIdx.x; i < n; i += stride) {
        out[i]     = hid[i];
        out[n + i] = x[i];
    }
}

// scatter first counts[b] rows of x back to soc positions
__global__ void scatter_kernel(const float* __restrict__ x,
                               const int* __restrict__ order,
                               const int* __restrict__ counts,
                               float* __restrict__ out_hid) {
    int row = blockIdx.x;
    int b = row / L_, i = row % L_;
    if (i >= counts[b]) return;
    const float* s = x + (long)row * D_;
    float* d = out_hid + ((long)b * L_ + order[b * L_ + i]) * D_;
    for (int c = threadIdx.x; c < D_; c += blockDim.x) d[c] = s[c];
}

// ---------------------------------------------------------------------------
static inline void launch_gemm(hipStream_t stream,
                               const _Float16* A, long aO, long aI,
                               const _Float16* Bt, long bO, long bI,
                               float* Cf, _Float16* Ch, long cO, long cI,
                               int innerCnt, int lda, int ldb, int ldc,
                               int M, int N, int K, int nz,
                               float scale, int accum, int gelu) {
    dim3 grid(N / 64, M / 128, nz);
    wmma_gemm_kernel<<<grid, dim3(256), 0, stream>>>(A, aO, aI, Bt, bO, bI,
                                                     Cf, Ch, cO, cI, innerCnt,
                                                     lda, ldb, ldc, K, scale, accum, gelu);
}

extern "C" void kernel_launch(void* const* d_in, const int* in_sizes, int n_in,
                              void* d_out, int out_size, void* d_ws, size_t ws_size,
                              hipStream_t stream) {
    (void)in_sizes; (void)n_in; (void)out_size; (void)ws_size;
    const float* hid = (const float*)d_in[0];
    const float* img = (const float*)d_in[1];
    const float* W[10] = {
        (const float*)d_in[2], (const float*)d_in[3], (const float*)d_in[4],
        (const float*)d_in[5], (const float*)d_in[6], (const float*)d_in[7],
        (const float*)d_in[8], (const float*)d_in[9], (const float*)d_in[10],
        (const float*)d_in[11]
    }; // Wq Wk Wv Wo Wq2 Wk2 Wv2 Wo2 W1 W2
    const int* seq = (const int*)d_in[12];
    const int* SOC = (const int*)d_in[13];
    const int* EOS = (const int*)d_in[14];

    const long BLD = (long)B_ * L_ * D_;

    char* ws = (char*)d_ws;
    size_t off = 0;
    auto alloc = [&](size_t bytes) -> char* {
        char* p = ws + off;
        off += (bytes + 255) & ~(size_t)255;
        return p;
    };

    float*    x     = (float*)alloc(BLD * 4);
    _Float16* xn    = (_Float16*)alloc(BLD * 2);
    _Float16* qf    = (_Float16*)alloc(BLD * 2);
    _Float16* kf    = (_Float16*)alloc(BLD * 2);
    _Float16* vf    = (_Float16*)alloc(BLD * 2);
    _Float16* vT    = (_Float16*)alloc(BLD * 2);
    _Float16* of    = (_Float16*)alloc(BLD * 2);
    _Float16* imgh  = (_Float16*)alloc(BLD * 2);
    _Float16* probs = (_Float16*)alloc((long)B_ * H_ * L_ * L_ * 2);
    _Float16* h1    = (_Float16*)alloc((long)B_ * L_ * DF_ * 2);
    _Float16* WT[10];
    const int WK[10] = {D_, D_, D_, D_, D_, D_, D_, D_, D_, DF_};
    const int WN[10] = {D_, D_, D_, D_, D_, D_, D_, D_, DF_, D_};
    for (int i = 0; i < 10; ++i) WT[i] = (_Float16*)alloc((long)WK[i] * WN[i] * 2);
    int* order  = (int*)alloc((long)B_ * L_ * 4);
    int* counts = (int*)alloc(B_ * 4);

    const int BL = B_ * L_;

    // --- compaction + input prep ---
    mask_order_kernel<<<1, 32, 0, stream>>>(seq, SOC, EOS, order, counts);
    gather_rows_kernel<<<BL, 128, 0, stream>>>(hid, order, counts, x);
    cvt_f16_kernel<<<2048, 256, 0, stream>>>(img, imgh, BLD);
    for (int i = 0; i < 10; ++i)
        transpose_w_kernel<<<1024, 256, 0, stream>>>(W[i], WT[i], WK[i], WN[i]);

    const long LL   = (long)L_ * L_;
    const long HLL  = (long)H_ * LL;
    const long LD   = (long)L_ * D_;
    const long DHL  = (long)DH_ * L_;
    const long HDHL = (long)H_ * DHL;

    // ================= block 1: masked self-attention =================
    ln_f16_kernel<<<BL, 256, 0, stream>>>(x, xn);
    launch_gemm(stream, xn,0,0, WT[0],0,0, nullptr,qf,0,0, 1, D_,D_,D_, BL,D_,D_, 1, 1.f,0,0);
    launch_gemm(stream, xn,0,0, WT[1],0,0, nullptr,kf,0,0, 1, D_,D_,D_, BL,D_,D_, 1, 1.f,0,0);
    launch_gemm(stream, xn,0,0, WT[2],0,0, nullptr,vf,0,0, 1, D_,D_,D_, BL,D_,D_, 1, 1.f,0,0);
    transpose_v_kernel<<<2048, 256, 0, stream>>>(vf, vT);
    // scores = q @ k^T / 8   (z = b*H + h)
    launch_gemm(stream, qf, LD, DH_, kf, LD, DH_, nullptr, probs, HLL, LL,
                H_, D_, D_, L_, L_, L_, DH_, B_ * H_, 0.125f, 0, 0);
    softmax_kernel<<<B_ * H_ * L_, 256, 0, stream>>>(probs, counts);
    // o = probs @ v
    launch_gemm(stream, probs, HLL, LL, vT, HDHL, DHL, nullptr, of, LD, DH_,
                H_, L_, L_, D_, L_, DH_, L_, B_ * H_, 1.f, 0, 0);
    // x += o @ Wo
    launch_gemm(stream, of,0,0, WT[3],0,0, x,nullptr,0,0, 1, D_,D_,D_, BL,D_,D_, 1, 1.f,1,0);

    // ================= block 2: cross-attention to img =================
    ln_f16_kernel<<<BL, 256, 0, stream>>>(x, xn);
    launch_gemm(stream, xn,  0,0, WT[4],0,0, nullptr,qf,0,0, 1, D_,D_,D_, BL,D_,D_, 1, 1.f,0,0);
    launch_gemm(stream, imgh,0,0, WT[5],0,0, nullptr,kf,0,0, 1, D_,D_,D_, BL,D_,D_, 1, 1.f,0,0);
    launch_gemm(stream, imgh,0,0, WT[6],0,0, nullptr,vf,0,0, 1, D_,D_,D_, BL,D_,D_, 1, 1.f,0,0);
    transpose_v_kernel<<<2048, 256, 0, stream>>>(vf, vT);
    launch_gemm(stream, qf, LD, DH_, kf, LD, DH_, nullptr, probs, HLL, LL,
                H_, D_, D_, L_, L_, L_, DH_, B_ * H_, 0.125f, 0, 0);
    softmax_kernel<<<B_ * H_ * L_, 256, 0, stream>>>(probs, nullptr);
    launch_gemm(stream, probs, HLL, LL, vT, HDHL, DHL, nullptr, of, LD, DH_,
                H_, L_, L_, D_, L_, DH_, L_, B_ * H_, 1.f, 0, 0);
    launch_gemm(stream, of,0,0, WT[7],0,0, x,nullptr,0,0, 1, D_,D_,D_, BL,D_,D_, 1, 1.f,1,0);

    // ================= block 3: FFN =================
    ln_f16_kernel<<<BL, 256, 0, stream>>>(x, xn);
    launch_gemm(stream, xn,0,0, WT[8],0,0, nullptr,h1,0,0, 1, D_,D_,DF_, BL,DF_,D_, 1, 1.f,0,/*gelu*/1);
    launch_gemm(stream, h1,0,0, WT[9],0,0, x,nullptr,0,0, 1, DF_,DF_,D_, BL,D_,DF_, 1, 1.f,1,0);

    // ================= outputs =================
    copy_out_kernel<<<4096, 256, 0, stream>>>(hid, x, (float*)d_out, BLD);
    scatter_kernel<<<BL, 128, 0, stream>>>(x, order, counts, (float*)d_out);
}
